// GraphSAGEClassifier_31310311588038
// MI455X (gfx1250) — matrix operand order repaired and checked by reading, not compiled, not run on hardware
//
#include <hip/hip_runtime.h>
#include <hip/hip_bf16.h>

typedef __attribute__((ext_vector_type(16))) _Float16 v16h;
typedef __attribute__((ext_vector_type(8)))  float    v8f;

#define NNODES   100000
#define DIM      64
#define NCLS     500
#define NCLS_PAD 512
#define NTILES   (NNODES / 16)   // 6250, exact

// ---------------------------------------------------------------------------
// WMMA helpers (CDNA5 16x16x32 f16 -> f32 accumulate)
// ---------------------------------------------------------------------------
__device__ __forceinline__ v8f wmma16(v16h a, v16h b, v8f c) {
    // (neg_a, A, neg_b, B, c_mod, C, reuse_a, reuse_b)
    return __builtin_amdgcn_wmma_f32_16x16x32_f16(false, a, false, b, (short)0, c,
                                                  false, false);
}

// A fragment: 16x32 f16 tile of a row-major f32 matrix [.., DIM] (scaled).
// Lane L: M = L&15; kb = 16*(L>=16); a[i]=A[M][K0+kb+i], i=0..15
__device__ __forceinline__ v16h load_a_frag(const float* __restrict__ src,
                                            int row0, int K0, int lane, float scale) {
    const int   m  = row0 + (lane & 15);
    const int   kb = K0 + ((lane >> 4) << 4);
    const float* p = src + (size_t)m * DIM + kb;
    v16h a;
#pragma unroll
    for (int i = 0; i < 8; ++i) a[i]     = (_Float16)(p[i] * scale);
#pragma unroll
    for (int i = 0; i < 8; ++i) a[8 + i] = (_Float16)(p[8 + i] * scale);
    return a;
}

// B fragment: 32x16 (KxN). out = h @ W^T  =>  B[k][n] = W[n][k], so lane L
// (column N = L&15) loads 16 contiguous f16 of row (n0+N) of W.
// Lane L: ks = K0 + 16*(L>=16); b[i] = W[n0+N][ks+i]
__device__ __forceinline__ v16h load_b_frag(const _Float16* __restrict__ w,
                                            int n0, int K0, int lane) {
    const int n  = n0 + (lane & 15);
    const int ks = K0 + ((lane >> 4) << 4);
    return *(const v16h*)(w + (size_t)n * DIM + ks);
}

// ---------------------------------------------------------------------------
// Convert all weights to f16 scratch: [Wl0 Wr0 Wl1 Wr1 Wl2 Wr2][Wc padded 512x64]
// ---------------------------------------------------------------------------
__global__ void cvt_weights_kernel(const float* __restrict__ wl0, const float* __restrict__ wr0,
                                   const float* __restrict__ wl1, const float* __restrict__ wr1,
                                   const float* __restrict__ wl2, const float* __restrict__ wr2,
                                   const float* __restrict__ wc, _Float16* __restrict__ w16) {
    const int t   = blockIdx.x * blockDim.x + threadIdx.x;
    const int WSZ = DIM * DIM;                 // 4096
    const int tot = 6 * WSZ + NCLS_PAD * DIM;  // 57344
    if (t >= tot) return;
    if (t < 6 * WSZ) {
        const int m = t / WSZ, i = t % WSZ;
        const float* src = (m == 0) ? wl0 : (m == 1) ? wr0 : (m == 2) ? wl1
                         : (m == 3) ? wr1 : (m == 4) ? wl2 : wr2;
        w16[t] = (_Float16)src[i];
    } else {
        const int i = t - 6 * WSZ;
        const int r = i / DIM;
        w16[t] = (_Float16)((r < NCLS) ? wc[i] : 0.0f);
    }
}

// ---------------------------------------------------------------------------
// Edge scatter: agg[dst,:] += h[src,:], deg[dst] += 1.  16 threads per edge,
// float4 gather (L2-resident) + 4 global_atomic_add_f32.
// ---------------------------------------------------------------------------
__global__ void scatter_kernel(const float* __restrict__ hin,
                               const long long* __restrict__ ei,
                               float* __restrict__ agg, float* __restrict__ deg,
                               int E) {
    const long long t = (long long)blockIdx.x * blockDim.x + threadIdx.x;
    const long long e = t >> 4;
    if (e >= E) return;
    const int c = (int)(t & 15);
    const long long s = ei[e];
    const long long d = ei[(long long)E + e];
    const float4 v = *(const float4*)(hin + s * DIM + c * 4);
    float* ap = agg + d * DIM + c * 4;
    atomicAdd(ap + 0, v.x);
    atomicAdd(ap + 1, v.y);
    atomicAdd(ap + 2, v.z);
    atomicAdd(ap + 3, v.w);
    if (c == 0) atomicAdd(deg + d, 1.0f);
}

// ---------------------------------------------------------------------------
// Fused SAGE layer: hout = [relu]( (agg/deg) @ Wl^T + bl + hin @ Wr^T )
// One wave per 16-row tile; 4 column tiles of 16; K=64 = 2 wmma steps.
// ---------------------------------------------------------------------------
__global__ void sage_gemm_kernel(const float* __restrict__ hin,
                                 const float* __restrict__ agg,
                                 const float* __restrict__ deg,
                                 const _Float16* __restrict__ wl,
                                 const _Float16* __restrict__ wr,
                                 const float* __restrict__ bl,
                                 float* __restrict__ hout, int relu) {
    const int lane = threadIdx.x & 31;
    const int wid  = threadIdx.x >> 5;
    const int tile = blockIdx.x * (blockDim.x >> 5) + wid;
    if (tile >= NTILES) return;
    const int row0 = tile * 16;

    const float dg   = deg[row0 + (lane & 15)];
    const float rdeg = 1.0f / fmaxf(dg, 1.0f);

    const v16h am0 = load_a_frag(agg, row0, 0,  lane, rdeg);
    const v16h am1 = load_a_frag(agg, row0, 32, lane, rdeg);
    const v16h ah0 = load_a_frag(hin, row0, 0,  lane, 1.0f);
    const v16h ah1 = load_a_frag(hin, row0, 32, lane, 1.0f);

    const int rbase = row0 + ((lane >> 4) << 3);
#pragma unroll
    for (int n0 = 0; n0 < DIM; n0 += 16) {
        v8f acc = {};
        acc = wmma16(am0, load_b_frag(wl, n0, 0,  lane), acc);
        acc = wmma16(am1, load_b_frag(wl, n0, 32, lane), acc);
        acc = wmma16(ah0, load_b_frag(wr, n0, 0,  lane), acc);
        acc = wmma16(ah1, load_b_frag(wr, n0, 32, lane), acc);

        const int   col  = n0 + (lane & 15);
        const float bias = bl[col];
#pragma unroll
        for (int c = 0; c < 8; ++c) {
            float v = acc[c] + bias;
            if (relu) v = fmaxf(v, 0.0f);
            hout[(size_t)(rbase + c) * DIM + col] = v;
        }
    }
}

// ---------------------------------------------------------------------------
// Row-wise L2 normalize, in place. One thread per node (fits in L2).
// ---------------------------------------------------------------------------
__global__ void l2norm_kernel(float* __restrict__ h) {
    const int n = blockIdx.x * blockDim.x + threadIdx.x;
    if (n >= NNODES) return;
    float* p = h + (size_t)n * DIM;
    float s = 0.0f;
#pragma unroll
    for (int i = 0; i < DIM / 4; ++i) {
        const float4 v = *(const float4*)(p + 4 * i);
        s += v.x * v.x + v.y * v.y + v.z * v.z + v.w * v.w;
    }
    const float inv = 1.0f / fmaxf(sqrtf(s), 1e-12f);
#pragma unroll
    for (int i = 0; i < DIM; ++i) p[i] *= inv;
}

// ---------------------------------------------------------------------------
// Classifier: out[N, 500] = hnorm @ Wc^T.  One wave per 16-row tile; A frags
// loaded once, 32 padded column tiles (Wc padded to 512 rows with zeros).
// ---------------------------------------------------------------------------
__global__ void classifier_kernel(const float* __restrict__ h,
                                  const _Float16* __restrict__ wc16,
                                  float* __restrict__ out) {
    const int lane = threadIdx.x & 31;
    const int wid  = threadIdx.x >> 5;
    const int tile = blockIdx.x * (blockDim.x >> 5) + wid;
    if (tile >= NTILES) return;
    const int row0 = tile * 16;

    const v16h a0 = load_a_frag(h, row0, 0,  lane, 1.0f);
    const v16h a1 = load_a_frag(h, row0, 32, lane, 1.0f);

    const int rbase = row0 + ((lane >> 4) << 3);
    for (int c0 = 0; c0 < NCLS_PAD; c0 += 16) {
        v8f acc = {};
        acc = wmma16(a0, load_b_frag(wc16, c0, 0,  lane), acc);
        acc = wmma16(a1, load_b_frag(wc16, c0, 32, lane), acc);
        const int col = c0 + (lane & 15);
        if (col < NCLS) {
#pragma unroll
            for (int c = 0; c < 8; ++c)
                out[(size_t)(rbase + c) * NCLS + col] = acc[c];
        }
    }
}

// ---------------------------------------------------------------------------
extern "C" void kernel_launch(void* const* d_in, const int* in_sizes, int n_in,
                              void* d_out, int out_size, void* d_ws, size_t ws_size,
                              hipStream_t stream) {
    const float*     x   = (const float*)d_in[0];
    const long long* ei  = (const long long*)d_in[1];
    const float*     Wl0 = (const float*)d_in[2];
    const float*     bl0 = (const float*)d_in[3];
    const float*     Wr0 = (const float*)d_in[4];
    const float*     Wl1 = (const float*)d_in[5];
    const float*     bl1 = (const float*)d_in[6];
    const float*     Wr1 = (const float*)d_in[7];
    const float*     Wl2 = (const float*)d_in[8];
    const float*     bl2 = (const float*)d_in[9];
    const float*     Wr2 = (const float*)d_in[10];
    const float*     Wc  = (const float*)d_in[11];
    const int E = in_sizes[1] / 2;

    // workspace carve (all chunks 32B-aligned)
    char* ws = (char*)d_ws;
    const size_t featB = (size_t)NNODES * DIM * sizeof(float);  // 25.6 MB
    float* hA  = (float*)ws;  ws += featB;
    float* hB  = (float*)ws;  ws += featB;
    float* agg = (float*)ws;  ws += featB;
    float* deg = (float*)ws;  ws += (size_t)NNODES * sizeof(float);
    _Float16* w16 = (_Float16*)ws;   // 6*4096 + 512*64 halfs

    // 1) weights -> f16 (Wc zero-padded to 512 rows)
    {
        const int tot = 6 * DIM * DIM + NCLS_PAD * DIM;
        cvt_weights_kernel<<<(tot + 255) / 256, 256, 0, stream>>>(
            Wl0, Wr0, Wl1, Wr1, Wl2, Wr2, Wc, w16);
    }

    // 2) three SAGE layers
    const float* biases[3] = {bl0, bl1, bl2};
    float*       outs[3]   = {hA, hB, hA};
    const float* hin = x;
    const int gemm_blocks = (NTILES + 7) / 8;       // 8 waves / block
    const long long sthreads = 16LL * E;
    const int sblocks = (int)((sthreads + 255) / 256);
    for (int l = 0; l < 3; ++l) {
        hipMemsetAsync(agg, 0, featB, stream);
        hipMemsetAsync(deg, 0, (size_t)NNODES * sizeof(float), stream);
        scatter_kernel<<<sblocks, 256, 0, stream>>>(hin, ei, agg, deg, E);
        const _Float16* wl = w16 + (size_t)l * 2 * DIM * DIM;
        const _Float16* wr = wl + DIM * DIM;
        sage_gemm_kernel<<<gemm_blocks, 256, 0, stream>>>(
            hin, agg, deg, wl, wr, biases[l], outs[l], (l < 2) ? 1 : 0);
        hin = outs[l];
    }

    // 3) L2 normalize (in place on hA) + classifier
    l2norm_kernel<<<(NNODES + 255) / 256, 256, 0, stream>>>(hA);
    classifier_kernel<<<gemm_blocks, 256, 0, stream>>>(
        hA, w16 + 6 * DIM * DIM, (float*)d_out);
}
